// DishFTGNN_82927228551440
// MI455X (gfx1250) — compile-verified, matrix operand order checked by reading, not run on hardware
//
#include <hip/hip_runtime.h>
#include <hip/hip_bf16.h>
#include <math.h>

// ---------------- problem constants ----------------
#define NNODES 50000
#define NEDGES 800000
#define FDIM   128
#define HIDD   128
#define NHEAD  8
#define NLAY   3
#define NGRAPH 64
#define NCLS   3
#define EPRIME (NEDGES + NNODES)          // edges + self loops = 850000
#define ROWTILES (NNODES / 16)            // 3125 (exact)

typedef __attribute__((ext_vector_type(16))) __bf16 v16bf;
typedef __attribute__((ext_vector_type(8)))  float  v8f;

struct Frag { union { v16bf v; uint4 q[2]; }; };

// ---------------- helpers ----------------
__device__ __forceinline__ unsigned short f2bf(float f) {
    unsigned int u = __float_as_uint(f);
    u += 0x7fffu + ((u >> 16) & 1u);      // round-to-nearest-even
    return (unsigned short)(u >> 16);
}
__device__ __forceinline__ float bf2f(unsigned short h) {
    return __uint_as_float(((unsigned int)h) << 16);
}
__device__ __forceinline__ void atomicMaxF(float* addr, float v) {
    if (v >= 0.0f) atomicMax((int*)addr, __float_as_int(v));
    else           atomicMin((unsigned int*)addr, (unsigned int)__float_as_int(v));
}

// ---------------- utility kernels ----------------
__global__ void fill_u32(unsigned int* __restrict__ p, unsigned int v, size_t n) {
    size_t i = (size_t)blockIdx.x * blockDim.x + threadIdx.x;
    if (i < n) p[i] = v;
}

__global__ void f32_to_bf16(const float* __restrict__ src,
                            unsigned short* __restrict__ dst, size_t n) {
    size_t i = (size_t)blockIdx.x * blockDim.x + threadIdx.x;
    if (i < n) dst[i] = f2bf(src[i]);
}

// Pack a KxN f32 matrix into the CDNA5 WMMA bf16 B-fragment layout:
// tileIdx = nt*KT + kt ; per tile 32 lanes x 16 halfs contiguous.
// lane&15 = N col, lane>>4 = half; e<8 -> K = 8*half+e ; e>=8 -> K = 16+8*half+(e-8)
__global__ void pack_b(const float* __restrict__ src, int ld, int trans,
                       int K, int Ncols, unsigned short* __restrict__ dst) {
    int tid = blockIdx.x * blockDim.x + threadIdx.x;
    if (tid >= K * Ncols) return;
    int e       = tid & 15;
    int lane    = (tid >> 4) & 31;
    int tileIdx = tid >> 9;
    int KT = K >> 5;
    int kt = tileIdx % KT;
    int nt = tileIdx / KT;
    int halfb = lane >> 4;
    int koff  = (e < 8) ? (8 * halfb + e) : (16 + 8 * halfb + (e - 8));
    int k = kt * 32 + koff;
    int n = nt * 16 + (lane & 15);
    float v = trans ? src[(size_t)n * ld + k] : src[(size_t)k * ld + n];
    dst[tid] = f2bf(v);
}

// -------- WMMA GEMM, 2 row-strips/wave, double-buffered B, LDS epilogue -----
// Optionally fuses the GAT attention dot products (alpha_src/dst) computed
// from the fp32 accumulators (ntiles must then be 8 heads x 8 tiles = 64).
// ntiles must be even.
__global__ void __launch_bounds__(128)
gemm_bf16_wmma(const unsigned short* __restrict__ A,
               const unsigned short* __restrict__ Bp,
               int ntiles, const float* __restrict__ bias, int relu,
               unsigned short* __restrict__ C, int ldc,
               const float* __restrict__ aSrc, const float* __restrict__ aDst,
               float* __restrict__ alS, float* __restrict__ alD,
               int nrowtiles) {
    __shared__ __align__(16) unsigned short stage[4][256];   // 512 B per wave
    int wave = threadIdx.x >> 5, lane = threadIdx.x & 31;
    int wu = blockIdx.x * (blockDim.x >> 5) + wave;
    int rt0 = wu * 2;
    if (rt0 >= nrowtiles) return;               // wave-uniform: EXEC all-1 at WMMA
    int rt1 = rt0 + 1;
    bool has1 = (rt1 < nrowtiles);
    int m = lane & 15, half = lane >> 4;
    int lrow = lane >> 1, lhalf = lane & 1;     // LDS read-back mapping
    unsigned short* st = stage[wave];

    // A fragments for both strips (two contiguous 8-half runs per lane)
    Frag a0[4], a1[4];
    size_t ar0 = (size_t)(rt0 * 16 + m) * 128;
#pragma unroll
    for (int kt = 0; kt < 4; ++kt) {
        a0[kt].q[0] = *(const uint4*)(A + ar0 + kt * 32 + 8 * half);
        a0[kt].q[1] = *(const uint4*)(A + ar0 + kt * 32 + 16 + 8 * half);
    }
    if (has1) {
        size_t ar1 = (size_t)(rt1 * 16 + m) * 128;
#pragma unroll
        for (int kt = 0; kt < 4; ++kt) {
            a1[kt].q[0] = *(const uint4*)(A + ar1 + kt * 32 + 8 * half);
            a1[kt].q[1] = *(const uint4*)(A + ar1 + kt * 32 + 16 + 8 * half);
        }
    }

    float ps0[8], pd0[8], ps1[8], pd1[8];
#pragma unroll
    for (int r = 0; r < 8; ++r) { ps0[r] = pd0[r] = ps1[r] = pd1[r] = 0.0f; }

    Frag b0[4], b1[4];
    auto loadB = [&](Frag* dst, int nt) {
#pragma unroll
        for (int kt = 0; kt < 4; ++kt) {
            const uint4* bp = (const uint4*)(Bp + (((size_t)nt * 4 + kt) * 32 + lane) * 16);
            dst[kt].q[0] = bp[0];
            dst[kt].q[1] = bp[1];
        }
    };

    auto process = [&](int nt, Frag* cur) {
        v8f c0 = {};
#pragma unroll
        for (int kt = 0; kt < 4; ++kt)
            c0 = __builtin_amdgcn_wmma_f32_16x16x32_bf16(
                     false, a0[kt].v, false, cur[kt].v, (short)0, c0, false, false);
        v8f c1 = {};
        if (has1) {
#pragma unroll
            for (int kt = 0; kt < 4; ++kt)
                c1 = __builtin_amdgcn_wmma_f32_16x16x32_bf16(
                         false, a1[kt].v, false, cur[kt].v, (short)0, c1, false, false);
        }

        int col = nt * 16 + m;
        float bv = bias ? bias[col] : 0.0f;

        // strip 0: bias/act, transpose through LDS, coalesced b128 store
#pragma unroll
        for (int r = 0; r < 8; ++r) {
            float v = c0[r] + bv;
            if (relu) v = fmaxf(v, 0.0f);
            c0[r] = v;
            st[(half * 8 + r) * 16 + m] = f2bf(v);
        }
        {
            uint4 pk = *(const uint4*)(st + lrow * 16 + lhalf * 8);
            *(uint4*)(C + (size_t)(rt0 * 16 + lrow) * ldc + nt * 16 + lhalf * 8) = pk;
        }
        if (has1) {
#pragma unroll
            for (int r = 0; r < 8; ++r) {
                float v = c1[r] + bv;
                if (relu) v = fmaxf(v, 0.0f);
                c1[r] = v;
                st[(half * 8 + r) * 16 + m] = f2bf(v);
            }
            uint4 pk = *(const uint4*)(st + lrow * 16 + lhalf * 8);
            *(uint4*)(C + (size_t)(rt1 * 16 + lrow) * ldc + nt * 16 + lhalf * 8) = pk;
        }

        // fused GAT attention dots (fp32 accumulators, pre-rounding)
        if (alS) {
            int hd = nt >> 3;
            float as = aSrc[hd * HIDD + (nt & 7) * 16 + m];
            float ad = aDst[hd * HIDD + (nt & 7) * 16 + m];
#pragma unroll
            for (int r = 0; r < 8; ++r) {
                ps0[r] += c0[r] * as; pd0[r] += c0[r] * ad;
                ps1[r] += c1[r] * as; pd1[r] += c1[r] * ad;
            }
            if ((nt & 7) == 7) {                 // head complete: reduce + write
#pragma unroll
                for (int r = 0; r < 8; ++r) {
                    for (int o = 1; o < 16; o <<= 1) {
                        ps0[r] += __shfl_xor(ps0[r], o);
                        pd0[r] += __shfl_xor(pd0[r], o);
                        ps1[r] += __shfl_xor(ps1[r], o);
                        pd1[r] += __shfl_xor(pd1[r], o);
                    }
                }
                if (m == 0) {
#pragma unroll
                    for (int r = 0; r < 8; ++r) {
                        int row0 = rt0 * 16 + half * 8 + r;
                        alS[row0 * NHEAD + hd] = ps0[r];
                        alD[row0 * NHEAD + hd] = pd0[r];
                        if (has1) {
                            int row1 = rt1 * 16 + half * 8 + r;
                            alS[row1 * NHEAD + hd] = ps1[r];
                            alD[row1 * NHEAD + hd] = pd1[r];
                        }
                    }
                }
#pragma unroll
                for (int r = 0; r < 8; ++r) { ps0[r] = pd0[r] = ps1[r] = pd1[r] = 0.0f; }
            }
        }
    };

    // software pipeline: ntiles is even (8 or 64); buffers have constant names
    loadB(b0, 0);
    for (int nt = 0; nt < ntiles; nt += 2) {
        loadB(b1, nt + 1);                       // in flight during wmma on b0
        process(nt, b0);
        if (nt + 2 < ntiles) loadB(b0, nt + 2);  // in flight during wmma on b1
        process(nt + 1, b1);
    }
}

// ---------------- edge softmax passes ----------------
__global__ void edge_max(const long long* __restrict__ ei,
                         const float* __restrict__ alS, const float* __restrict__ alD,
                         float* __restrict__ emax) {
    int e = blockIdx.x * blockDim.x + threadIdx.x;
    if (e >= EPRIME) return;
    long long s, d;
    if (e < NEDGES) { s = ei[e]; d = ei[NEDGES + e]; } else { s = d = e - NEDGES; }
#pragma unroll
    for (int h = 0; h < NHEAD; ++h) {
        float v = alS[s * NHEAD + h] + alD[d * NHEAD + h];
        v = (v > 0.0f) ? v : 0.2f * v;
        atomicMaxF(&emax[d * NHEAD + h], v);
    }
}

__global__ void edge_exp(const long long* __restrict__ ei,
                         const float* __restrict__ alS, const float* __restrict__ alD,
                         const float* __restrict__ emax,
                         float* __restrict__ aexp, float* __restrict__ denom) {
    int e = blockIdx.x * blockDim.x + threadIdx.x;
    if (e >= EPRIME) return;
    long long s, d;
    if (e < NEDGES) { s = ei[e]; d = ei[NEDGES + e]; } else { s = d = e - NEDGES; }
#pragma unroll
    for (int h = 0; h < NHEAD; ++h) {
        float v = alS[s * NHEAD + h] + alD[d * NHEAD + h];
        v = (v > 0.0f) ? v : 0.2f * v;
        float a = __expf(v - emax[d * NHEAD + h]);
        aexp[(size_t)e * NHEAD + h] = a;
        atomicAdd(&denom[d * NHEAD + h], a);
    }
}

// one block (128 thr) per edge; head coefficients staged through LDS; bf16 xl
// gather (102 MB layer buffer -> resident in 192 MB L2), fp32 atomic scatter.
__global__ void edge_msg(const long long* __restrict__ ei,
                         const unsigned short* __restrict__ XL,
                         const float* __restrict__ aexp, const float* __restrict__ denom,
                         float* __restrict__ acc) {
    __shared__ float cf[NHEAD];
    int e = blockIdx.x;
    long long s, d;
    if (e < NEDGES) { s = ei[e]; d = ei[NEDGES + e]; } else { s = d = e - NEDGES; }
    int t = threadIdx.x;
    if (t < NHEAD) cf[t] = aexp[(size_t)e * NHEAD + t] / (denom[d * NHEAD + t] + 1e-16f);
    __syncthreads();
    const unsigned short* x = XL + (size_t)s * (NHEAD * HIDD);
    float sum = 0.0f;
#pragma unroll
    for (int h = 0; h < NHEAD; ++h) sum += cf[h] * bf2f(x[h * HIDD + t]);
    atomicAdd(&acc[(size_t)d * HIDD + t], sum);
}

__global__ void gat_finalize(const float* __restrict__ acc, const float* __restrict__ bias,
                             int relu, unsigned short* __restrict__ Hbf) {
    size_t i = (size_t)blockIdx.x * blockDim.x + threadIdx.x;
    if (i >= (size_t)NNODES * HIDD) return;
    int c = i & (HIDD - 1);
    float v = acc[i] * (1.0f / NHEAD) + bias[c];
    if (relu) v = fmaxf(v, 0.0f);
    Hbf[i] = f2bf(v);
}

// ---------------- fused GRU cell ----------------
// Per 16-row strip: 6 fp32 WMMA accumulators (r/z/n x ih/hh), gates in regs,
// gi/gh never hit memory.
__global__ void __launch_bounds__(128)
gru_fused(const unsigned short* __restrict__ Hbf,
          const unsigned short* __restrict__ Mbf,
          const float* __restrict__ memF,
          const unsigned short* __restrict__ pWih,
          const unsigned short* __restrict__ pWhh,
          const float* __restrict__ bih, const float* __restrict__ bhh,
          float* __restrict__ Hout, int nrowtiles) {
    int wave = threadIdx.x >> 5, lane = threadIdx.x & 31;
    int rt = blockIdx.x * (blockDim.x >> 5) + wave;
    if (rt >= nrowtiles) return;
    int m = lane & 15, half = lane >> 4;
    size_t arow = (size_t)(rt * 16 + m) * 128;

    Frag ah[4], am[4];
#pragma unroll
    for (int kt = 0; kt < 4; ++kt) {
        ah[kt].q[0] = *(const uint4*)(Hbf + arow + kt * 32 + 8 * half);
        ah[kt].q[1] = *(const uint4*)(Hbf + arow + kt * 32 + 16 + 8 * half);
        am[kt].q[0] = *(const uint4*)(Mbf + arow + kt * 32 + 8 * half);
        am[kt].q[1] = *(const uint4*)(Mbf + arow + kt * 32 + 16 + 8 * half);
    }
    for (int ct = 0; ct < 8; ++ct) {
        v8f ir = {}, iz = {}, inn = {}, hr = {}, hz = {}, hn = {};
#pragma unroll
        for (int kt = 0; kt < 4; ++kt) {
            Frag b;
            const uint4* bp;
            bp = (const uint4*)(pWih + (((size_t)(ct)*4 + kt) * 32 + lane) * 16);
            b.q[0] = bp[0]; b.q[1] = bp[1];
            ir = __builtin_amdgcn_wmma_f32_16x16x32_bf16(false, ah[kt].v, false, b.v, (short)0, ir, false, false);
            bp = (const uint4*)(pWih + (((size_t)(8 + ct) * 4 + kt) * 32 + lane) * 16);
            b.q[0] = bp[0]; b.q[1] = bp[1];
            iz = __builtin_amdgcn_wmma_f32_16x16x32_bf16(false, ah[kt].v, false, b.v, (short)0, iz, false, false);
            bp = (const uint4*)(pWih + (((size_t)(16 + ct) * 4 + kt) * 32 + lane) * 16);
            b.q[0] = bp[0]; b.q[1] = bp[1];
            inn = __builtin_amdgcn_wmma_f32_16x16x32_bf16(false, ah[kt].v, false, b.v, (short)0, inn, false, false);
            bp = (const uint4*)(pWhh + (((size_t)(ct)*4 + kt) * 32 + lane) * 16);
            b.q[0] = bp[0]; b.q[1] = bp[1];
            hr = __builtin_amdgcn_wmma_f32_16x16x32_bf16(false, am[kt].v, false, b.v, (short)0, hr, false, false);
            bp = (const uint4*)(pWhh + (((size_t)(8 + ct) * 4 + kt) * 32 + lane) * 16);
            b.q[0] = bp[0]; b.q[1] = bp[1];
            hz = __builtin_amdgcn_wmma_f32_16x16x32_bf16(false, am[kt].v, false, b.v, (short)0, hz, false, false);
            bp = (const uint4*)(pWhh + (((size_t)(16 + ct) * 4 + kt) * 32 + lane) * 16);
            b.q[0] = bp[0]; b.q[1] = bp[1];
            hn = __builtin_amdgcn_wmma_f32_16x16x32_bf16(false, am[kt].v, false, b.v, (short)0, hn, false, false);
        }
        int col = ct * 16 + m;
        float bri = bih[col], bzi = bih[128 + col], bni = bih[256 + col];
        float brh = bhh[col], bzh = bhh[128 + col], bnh = bhh[256 + col];
#pragma unroll
        for (int r = 0; r < 8; ++r) {
            int rowg = rt * 16 + half * 8 + r;
            float mv = memF[(size_t)rowg * 128 + col];
            float rg = 1.0f / (1.0f + __expf(-(ir[r] + bri + hr[r] + brh)));
            float zg = 1.0f / (1.0f + __expf(-(iz[r] + bzi + hz[r] + bzh)));
            float ng = tanhf(inn[r] + bni + rg * (hn[r] + bnh));
            Hout[(size_t)rowg * 128 + col] = (1.0f - zg) * ng + zg * mv;
        }
    }
}

// ---------------- pooling + classifier ----------------
__global__ void pool_accum(const float* __restrict__ H, const long long* __restrict__ batch,
                           float* __restrict__ gsum, float* __restrict__ gcnt) {
    size_t i = (size_t)blockIdx.x * blockDim.x + threadIdx.x;
    if (i >= (size_t)NNODES * HIDD) return;
    int n = (int)(i >> 7), c = (int)(i & 127);
    long long g = batch[n];
    atomicAdd(&gsum[g * HIDD + c], H[i]);
    if (c == 0) atomicAdd(&gcnt[g], 1.0f);
}

__global__ void pool_final(const float* __restrict__ gsum, const float* __restrict__ gcnt,
                           float* __restrict__ pooled, float* __restrict__ out_pooled) {
    int i = blockIdx.x * blockDim.x + threadIdx.x;
    if (i >= NGRAPH * HIDD) return;
    float v = gsum[i] / fmaxf(gcnt[i >> 7], 1.0f);
    pooled[i] = v;
    out_pooled[i] = v;
}

__global__ void classifier(const float* __restrict__ pooled,
                           const float* __restrict__ W1, const float* __restrict__ b1,
                           const float* __restrict__ W2, const float* __restrict__ b2,
                           float* __restrict__ out_logits, float* __restrict__ out_preds) {
    int g = threadIdx.x;
    if (g >= NGRAPH) return;
    const float* p = pooled + (size_t)g * HIDD;
    float lg[NCLS];
#pragma unroll
    for (int cc = 0; cc < NCLS; ++cc) lg[cc] = b2[cc];
    for (int j = 0; j < 64; ++j) {
        float t = b1[j];
        for (int k = 0; k < 128; ++k) t += p[k] * W1[k * 64 + j];
        t = fmaxf(t, 0.0f);
#pragma unroll
        for (int cc = 0; cc < NCLS; ++cc) lg[cc] += t * W2[j * NCLS + cc];
    }
    float mx = fmaxf(lg[0], fmaxf(lg[1], lg[2]));
    float e0 = __expf(lg[0] - mx), e1 = __expf(lg[1] - mx), e2 = __expf(lg[2] - mx);
    float inv = 1.0f / (e0 + e1 + e2);
    out_logits[g * 3 + 0] = lg[0]; out_logits[g * 3 + 1] = lg[1]; out_logits[g * 3 + 2] = lg[2];
    out_preds[g * 3 + 0] = e0 * inv; out_preds[g * 3 + 1] = e1 * inv; out_preds[g * 3 + 2] = e2 * inv;
}

// ---------------- host launch ----------------
extern "C" void kernel_launch(void* const* d_in, const int* in_sizes, int n_in,
                              void* d_out, int out_size, void* d_ws, size_t ws_size,
                              hipStream_t stream) {
    const float*     x      = (const float*)d_in[0];
    const long long* ei     = (const long long*)d_in[1];
    const long long* batch  = (const long long*)d_in[2];
    const float*     memory = (const float*)d_in[3];
    const float* enc_W = (const float*)d_in[4];
    const float* enc_b = (const float*)d_in[5];
    const float* gat_W = (const float*)d_in[6];
    const float* aS    = (const float*)d_in[7];
    const float* aD    = (const float*)d_in[8];
    const float* gat_b = (const float*)d_in[9];
    const float* Wih   = (const float*)d_in[10];
    const float* Whh   = (const float*)d_in[11];
    const float* bih   = (const float*)d_in[12];
    const float* bhh   = (const float*)d_in[13];
    const float* c1W   = (const float*)d_in[14];
    const float* c1b   = (const float*)d_in[15];
    const float* c2W   = (const float*)d_in[16];
    const float* c2b   = (const float*)d_in[17];
    float* out = (float*)d_out;

    char* ws = (char*)d_ws;
    size_t off = 0;
    auto take = [&](size_t bytes) -> char* {
        char* p = ws + off;
        off = (off + bytes + 255) & ~(size_t)255;
        return p;
    };

    unsigned short* pW_enc = (unsigned short*)take((size_t)128 * 128 * 2);
    unsigned short* pW_gat = (unsigned short*)take((size_t)NLAY * 128 * 1024 * 2);
    unsigned short* pW_ih  = (unsigned short*)take((size_t)128 * 384 * 2);
    unsigned short* pW_hh  = (unsigned short*)take((size_t)128 * 384 * 2);
    unsigned short* x_bf   = (unsigned short*)take((size_t)NNODES * 128 * 2);
    unsigned short* h_bf   = (unsigned short*)take((size_t)NNODES * 128 * 2);
    unsigned short* mem_bf = (unsigned short*)take((size_t)NNODES * 128 * 2);
    unsigned short* xl_bf  = (unsigned short*)take((size_t)NNODES * 1024 * 2);
    float* alS   = (float*)take((size_t)NNODES * NHEAD * 4);
    float* alD   = (float*)take((size_t)NNODES * NHEAD * 4);
    float* emax  = (float*)take((size_t)NNODES * NHEAD * 4);
    float* denom = (float*)take((size_t)NNODES * NHEAD * 4);
    float* aexp  = (float*)take((size_t)EPRIME * NHEAD * 4);
    float* acc   = (float*)take((size_t)NNODES * 128 * 4);
    float* hfin  = (float*)take((size_t)NNODES * 128 * 4);
    float* gsum  = (float*)take((size_t)NGRAPH * 128 * 4);
    float* gcnt  = (float*)take((size_t)NGRAPH * 4);
    float* pooled = (float*)take((size_t)NGRAPH * 128 * 4);

    auto cdiv = [](long long a, long long b) { return (int)((a + b - 1) / b); };

    // --- pack weights into WMMA B-fragment layout (bf16) ---
    pack_b<<<cdiv(128 * 128, 256), 256, 0, stream>>>(enc_W, 128, 0, 128, 128, pW_enc);
    for (int l = 0; l < NLAY; ++l)
        pack_b<<<cdiv(128 * 1024, 256), 256, 0, stream>>>(
            gat_W + (size_t)l * 128 * 1024, 1024, 0, 128, 1024,
            pW_gat + (size_t)l * 128 * 1024);
    pack_b<<<cdiv(128 * 384, 256), 256, 0, stream>>>(Wih, 128, 1, 128, 384, pW_ih);
    pack_b<<<cdiv(128 * 384, 256), 256, 0, stream>>>(Whh, 128, 1, 128, 384, pW_hh);

    // --- fp32 -> bf16 activations ---
    f32_to_bf16<<<cdiv((size_t)NNODES * 128, 256), 256, 0, stream>>>(x, x_bf, (size_t)NNODES * 128);
    f32_to_bf16<<<cdiv((size_t)NNODES * 128, 256), 256, 0, stream>>>(memory, mem_bf, (size_t)NNODES * 128);

    const int WU = (ROWTILES + 1) / 2;          // wave-units (2 row tiles each)

    // --- encoder: h = relu(x @ enc_W + b) ---
    gemm_bf16_wmma<<<cdiv(WU, 4), 128, 0, stream>>>(
        x_bf, pW_enc, 8, enc_b, 1, h_bf, 128,
        nullptr, nullptr, nullptr, nullptr, ROWTILES);

    // --- GAT layers ---
    for (int l = 0; l < NLAY; ++l) {
        // lin + fused alpha_src/alpha_dst
        gemm_bf16_wmma<<<cdiv(WU, 4), 128, 0, stream>>>(
            h_bf, pW_gat + (size_t)l * 128 * 1024, 64, nullptr, 0, xl_bf, 1024,
            aS + (size_t)l * NHEAD * HIDD, aD + (size_t)l * NHEAD * HIDD,
            alS, alD, ROWTILES);

        fill_u32<<<cdiv((size_t)NNODES * NHEAD, 256), 256, 0, stream>>>(
            (unsigned int*)emax, 0xFF800000u, (size_t)NNODES * NHEAD);      // -inf
        fill_u32<<<cdiv((size_t)NNODES * NHEAD, 256), 256, 0, stream>>>(
            (unsigned int*)denom, 0u, (size_t)NNODES * NHEAD);
        fill_u32<<<cdiv((size_t)NNODES * 128, 256), 256, 0, stream>>>(
            (unsigned int*)acc, 0u, (size_t)NNODES * 128);

        edge_max<<<cdiv(EPRIME, 256), 256, 0, stream>>>(ei, alS, alD, emax);
        edge_exp<<<cdiv(EPRIME, 256), 256, 0, stream>>>(ei, alS, alD, emax, aexp, denom);
        edge_msg<<<EPRIME, 128, 0, stream>>>(ei, xl_bf, aexp, denom, acc);

        gat_finalize<<<cdiv((size_t)NNODES * 128, 256), 256, 0, stream>>>(
            acc, gat_b + (size_t)l * HIDD, (l < NLAY - 1) ? 1 : 0, h_bf);
    }

    // --- fused GRU cell ---
    gru_fused<<<cdiv(ROWTILES, 4), 128, 0, stream>>>(
        h_bf, mem_bf, memory, pW_ih, pW_hh, bih, bhh, hfin, ROWTILES);

    // --- global mean pool ---
    fill_u32<<<cdiv(NGRAPH * 128, 256), 256, 0, stream>>>((unsigned int*)gsum, 0u, NGRAPH * 128);
    fill_u32<<<1, 64, 0, stream>>>((unsigned int*)gcnt, 0u, NGRAPH);
    pool_accum<<<cdiv((size_t)NNODES * 128, 256), 256, 0, stream>>>(hfin, batch, gsum, gcnt);
    pool_final<<<cdiv(NGRAPH * 128, 256), 256, 0, stream>>>(gsum, gcnt, pooled, out + NGRAPH * NCLS);

    // --- classifier + softmax (tiny, fp32) ---
    classifier<<<1, 64, 0, stream>>>(pooled, c1W, c1b, c2W, c2b,
                                     out, out + NGRAPH * NCLS + NGRAPH * HIDD);
}
